// Net_17351667876196
// MI455X (gfx1250) — matrix operand order repaired and checked by reading, not compiled, not run on hardware
//
#include <hip/hip_runtime.h>
#include <hip/hip_bf16.h>

// ---------------------------------------------------------------------------
// GCN (DGL GraphConv norm='both') on MI455X / gfx1250.
// GEMMs via v_wmma_f32_16x16x32_bf16 (bf16 in, f32 accumulate), 4 accumulators
// per wave (16x64 tile) reusing each A fetch across 4 WMMAs.
// Aggregation via edge-parallel waves + native f32 global atomics (L2-resident:
// the 102.4MB per-layer node table fits the 192MB L2).
// ---------------------------------------------------------------------------

typedef __attribute__((ext_vector_type(16))) __bf16 v16bf;
typedef __attribute__((ext_vector_type(8)))  float  v8f;

#define NNODES 100000
#define NEDGES 3200000
#define F_IN   767
#define K1PAD  768     // 767 padded to multiple of 32
#define HID    256
#define NOUT   10

union V16U { v16bf v; uint4 q[2]; };

// -------------------------------- utility ----------------------------------

__global__ void fill_zero_f32(float* __restrict__ p, long n) {
  long i = (long)blockIdx.x * blockDim.x + threadIdx.x;
  if (i < n) p[i] = 0.0f;
}

// degree accumulation (float counters -> native global_atomic_add_f32)
__global__ void degree_kernel(const int* __restrict__ src, const int* __restrict__ dst,
                              float* __restrict__ outdeg, float* __restrict__ indeg, int ne) {
  int i = blockIdx.x * blockDim.x + threadIdx.x;
  if (i < ne) {
    unsafeAtomicAdd(&outdeg[src[i]], 1.0f);
    unsafeAtomicAdd(&indeg[dst[i]],  1.0f);
  }
}

// deg -> clip(deg,1)^-0.5, in place for both norms
__global__ void norm_kernel(float* __restrict__ outn, float* __restrict__ inn, int n) {
  int i = blockIdx.x * blockDim.x + threadIdx.x;
  if (i < n) {
    outn[i] = rsqrtf(fmaxf(outn[i], 1.0f));
    inn[i]  = rsqrtf(fmaxf(inn[i],  1.0f));
  }
}

// ---------------------- weight packing to WMMA B layout --------------------
// B (32x16 bf16 per k-tile) lane layout: elements 0..7 hold K=kt*32+base+0..7,
// elements 8..15 hold K=kt*32+base+16..23, base = (lane<16 ? 0 : 8); column N = nt*16 + lane%16.
// Packed element offset = ((kt*numNT + nt)*32 + lane)*16 + e  (contiguous 32B
// per lane -> two uint4 loads in the GEMM).
__global__ void pack_w_kernel(const float* __restrict__ W, __bf16* __restrict__ Wp,
                              int Kreal, int Nreal, int Kpad, int Npad) {
  int numNT = Npad >> 4;
  int total = (Kpad >> 5) * numNT * 32;
  int t = blockIdx.x * blockDim.x + threadIdx.x;
  if (t >= total) return;
  int lane = t & 31;
  int tile = t >> 5;
  int nt = tile % numNT;
  int kt = tile / numNT;
  int kbase = kt * 32 + ((lane & 16) ? 8 : 0);
  int col   = nt * 16 + (lane & 15);
  __bf16* dstp = Wp + (size_t)t * 16;
#pragma unroll
  for (int e = 0; e < 16; ++e) {
    int k = kbase + (e < 8 ? e : e + 8);
    float v = (k < Kreal && col < Nreal) ? W[(size_t)k * Nreal + col] : 0.0f;
    dstp[e] = (__bf16)v;
  }
}

// layer-1 input: Abf[row][k] = bf16(features[row][k] * out_norm[row]), K padded
__global__ void pre_kernel(const float* __restrict__ feats, const float* __restrict__ outn,
                           __bf16* __restrict__ A, long total) {
  long i = (long)blockIdx.x * blockDim.x + threadIdx.x;
  if (i >= total) return;
  int row = (int)(i / K1PAD);
  int k   = (int)(i % K1PAD);
  float v = (k < F_IN) ? feats[(size_t)row * F_IN + k] * outn[row] : 0.0f;
  A[i] = (__bf16)v;
}

// ------------------------------- WMMA GEMM ---------------------------------
// A per-lane layout (ISA 7.12.2, 16-bit A 16x32): row = mt*16 + lane%16,
// elements 0..7 -> K = k0+base+0..7, elements 8..15 -> K = k0+base+16..23,
// base = (lane<16 ? 0 : 8)  => two aligned 16B loads per K-step.
// One wave computes a 16x64 f32 tile (4 accumulators), reusing the A fetch:
// per K-step: 2 A-loads + 8 B-loads feed 4 WMMAs.
__global__ void gemm4_kernel(const __bf16* __restrict__ A, int ldA,
                             const __bf16* __restrict__ Wp, float* __restrict__ C,
                             int K, int N, int numNT, int numWaveTiles) {
  int wave = (int)((blockIdx.x * (long)blockDim.x + threadIdx.x) >> 5);
  int lane = threadIdx.x & 31;
  if (wave >= numWaveTiles) return;          // wave-uniform guard: EXEC all-1 for WMMA
  const int ntGroups = numNT >> 2;
  const int mt  = wave / ntGroups;
  const int nt0 = (wave % ntGroups) << 2;
  const int half = lane >> 4;
  const int r    = lane & 15;

  v8f c0 = {}, c1 = {}, c2 = {}, c3 = {};
  const __bf16* arow = A + (size_t)(mt * 16 + r) * ldA + half * 8;
  const __bf16* bptr = Wp + ((size_t)nt0 * 32 + lane) * 16;
  const long bstep = (long)numNT * 32 * 16;  // elements per k-tile row of subtiles
  const long bsub  = (long)32 * 16;          // 512 elements per n-subtile

  for (int k0 = 0; k0 < K; k0 += 32) {
    V16U a, b0, b1, b2, b3;
    a.q[0] = *(const uint4*)(arow + k0);
    a.q[1] = *(const uint4*)(arow + k0 + 16);
    b0.q[0] = *(const uint4*)(bptr + 0 * bsub);
    b0.q[1] = *(const uint4*)(bptr + 0 * bsub + 8);
    b1.q[0] = *(const uint4*)(bptr + 1 * bsub);
    b1.q[1] = *(const uint4*)(bptr + 1 * bsub + 8);
    b2.q[0] = *(const uint4*)(bptr + 2 * bsub);
    b2.q[1] = *(const uint4*)(bptr + 2 * bsub + 8);
    b3.q[0] = *(const uint4*)(bptr + 3 * bsub);
    b3.q[1] = *(const uint4*)(bptr + 3 * bsub + 8);
    bptr += bstep;
    c0 = __builtin_amdgcn_wmma_f32_16x16x32_bf16(false, a.v, false, b0.v, (short)0, c0, false, false);
    c1 = __builtin_amdgcn_wmma_f32_16x16x32_bf16(false, a.v, false, b1.v, (short)0, c1, false, false);
    c2 = __builtin_amdgcn_wmma_f32_16x16x32_bf16(false, a.v, false, b2.v, (short)0, c2, false, false);
    c3 = __builtin_amdgcn_wmma_f32_16x16x32_bf16(false, a.v, false, b3.v, (short)0, c3, false, false);
  }

  // C layout: element rr -> row mt*16 + (lane<16?0:8)+rr, col nt*16 + lane%16
  const int row0 = mt * 16 + (half << 3);
  const int colr = lane & 15;
#pragma unroll
  for (int rr = 0; rr < 8; ++rr) {
    float* crow = C + (size_t)(row0 + rr) * N + colr;
    crow[(nt0 + 0) * 16] = c0[rr];
    crow[(nt0 + 1) * 16] = c1[rr];
    crow[(nt0 + 2) * 16] = c2[rr];
    crow[(nt0 + 3) * 16] = c3[rr];
  }
}

// final 256->10 GEMM with bias, N padded to 16 in Wp, masked store
__global__ void gemm_final_kernel(const __bf16* __restrict__ A, int ldA,
                                  const __bf16* __restrict__ Wp,
                                  const float* __restrict__ bias,
                                  float* __restrict__ out, int K, int numTiles) {
  int wave = (int)((blockIdx.x * (long)blockDim.x + threadIdx.x) >> 5);
  int lane = threadIdx.x & 31;
  if (wave >= numTiles) return;
  const int half = lane >> 4;
  const int r    = lane & 15;
  v8f c = {};
  const __bf16* arow = A + (size_t)(wave * 16 + r) * ldA + half * 8;
  const __bf16* bptr = Wp + (size_t)lane * 16;
  for (int k0 = 0; k0 < K; k0 += 32) {
    V16U a, b;
    a.q[0] = *(const uint4*)(arow + k0);
    a.q[1] = *(const uint4*)(arow + k0 + 16);
    b.q[0] = *(const uint4*)(bptr);
    b.q[1] = *(const uint4*)(bptr + 8);
    bptr += 32 * 16;
    c = __builtin_amdgcn_wmma_f32_16x16x32_bf16(false, a.v, false, b.v, (short)0, c, false, false);
  }
  int col  = lane & 15;
  int row0 = wave * 16 + (half << 3);
  if (col < NOUT) {
    float b = bias[col];
#pragma unroll
    for (int rr = 0; rr < 8; ++rr)
      out[(size_t)(row0 + rr) * NOUT + col] = c[rr] + b;
  }
}

// --------------------------- edge aggregation ------------------------------
// One wave per edge: each lane reads 32B of the 1KB source row with two b128
// loads (wave covers the row fully coalesced), then 8 native f32 atomic adds.
__global__ void agg_kernel(const int* __restrict__ src, const int* __restrict__ dst,
                           const float* __restrict__ H, float* __restrict__ AGG, int ne) {
  long gid = (long)blockIdx.x * blockDim.x + threadIdx.x;
  int e    = (int)(gid >> 5);
  int lane = (int)(gid & 31);
  if (e >= ne) return;
  int s = src[e];
  int d = dst[e];
  const float4* hs = (const float4*)(H + (size_t)s * HID) + lane * 2;
  float4 x0 = hs[0];
  float4 x1 = hs[1];
  float* ad = AGG + (size_t)d * HID + lane * 8;
  unsafeAtomicAdd(ad + 0, x0.x);
  unsafeAtomicAdd(ad + 1, x0.y);
  unsafeAtomicAdd(ad + 2, x0.z);
  unsafeAtomicAdd(ad + 3, x0.w);
  unsafeAtomicAdd(ad + 4, x1.x);
  unsafeAtomicAdd(ad + 5, x1.y);
  unsafeAtomicAdd(ad + 6, x1.z);
  unsafeAtomicAdd(ad + 7, x1.w);
}

// fused: relu(agg*in_norm + b) [* out_norm] -> bf16 activation buffer
__global__ void post_kernel(const float* __restrict__ AGG, const float* __restrict__ inn,
                            const float* __restrict__ outn, const float* __restrict__ bias,
                            __bf16* __restrict__ Aout, int scaleOut, long total) {
  long i = (long)blockIdx.x * blockDim.x + threadIdx.x;
  if (i >= total) return;
  int row = (int)(i >> 8);          // HID == 256
  int j   = (int)(i & 255);
  float v = AGG[i] * inn[row] + bias[j];
  v = fmaxf(v, 0.0f);
  if (scaleOut) v *= outn[row];
  Aout[i] = (__bf16)v;
}

// ------------------------------- launcher ----------------------------------

extern "C" void kernel_launch(void* const* d_in, const int* in_sizes, int n_in,
                              void* d_out, int out_size, void* d_ws, size_t ws_size,
                              hipStream_t stream) {
  const float* feats = (const float*)d_in[0];
  const int*   esrc  = (const int*)  d_in[1];
  const int*   edst  = (const int*)  d_in[2];
  const float* W1 = (const float*)d_in[3];  const float* b1 = (const float*)d_in[4];
  const float* W2 = (const float*)d_in[5];  const float* b2 = (const float*)d_in[6];
  const float* W3 = (const float*)d_in[7];  const float* b3 = (const float*)d_in[8];
  const float* Wf = (const float*)d_in[9];  const float* bfv = (const float*)d_in[10];
  float* out = (float*)d_out;

  char* ws = (char*)d_ws;
  size_t off = 0;
  auto alloc = [&](size_t bytes) -> void* {
    void* p = ws + off;
    off += (bytes + 255) & ~(size_t)255;
    return p;
  };
  float*  outn = (float*) alloc((size_t)NNODES * 4);
  float*  inn  = (float*) alloc((size_t)NNODES * 4);
  __bf16* Abf  = (__bf16*)alloc((size_t)NNODES * K1PAD * 2);   // also reused @ld=256
  float*  H    = (float*) alloc((size_t)NNODES * HID * 4);
  float*  AGG  = (float*) alloc((size_t)NNODES * HID * 4);
  __bf16* W1p  = (__bf16*)alloc((size_t)K1PAD * HID * 2);
  __bf16* W2p  = (__bf16*)alloc((size_t)HID * HID * 2);
  __bf16* W3p  = (__bf16*)alloc((size_t)HID * HID * 2);
  __bf16* Wfp  = (__bf16*)alloc((size_t)HID * 16 * 2);

  const int BT = 256;
  auto blocks = [](long n, int bt) { return (int)((n + bt - 1) / bt); };

  // 1) degrees -> norms
  fill_zero_f32<<<blocks(NNODES, BT), BT, 0, stream>>>(outn, NNODES);
  fill_zero_f32<<<blocks(NNODES, BT), BT, 0, stream>>>(inn,  NNODES);
  degree_kernel<<<blocks(NEDGES, BT), BT, 0, stream>>>(esrc, edst, outn, inn, NEDGES);
  norm_kernel<<<blocks(NNODES, BT), BT, 0, stream>>>(outn, inn, NNODES);

  // 2) pack weights into WMMA-native bf16 layout
  pack_w_kernel<<<blocks((long)(K1PAD/32)*(HID/16)*32, BT), BT, 0, stream>>>(W1, W1p, F_IN, HID, K1PAD, HID);
  pack_w_kernel<<<blocks((long)(HID/32)*(HID/16)*32, BT), BT, 0, stream>>>(W2, W2p, HID, HID, HID, HID);
  pack_w_kernel<<<blocks((long)(HID/32)*(HID/16)*32, BT), BT, 0, stream>>>(W3, W3p, HID, HID, HID, HID);
  pack_w_kernel<<<blocks((long)(HID/32)*1*32, BT), BT, 0, stream>>>(Wf, Wfp, HID, NOUT, HID, 16);

  // 3) layer-1 input: scale by out_norm, cast bf16, pad K to 768
  const long preTotal = (long)NNODES * K1PAD;
  pre_kernel<<<blocks(preTotal, BT), BT, 0, stream>>>(feats, outn, Abf, preTotal);

  const int  mtiles     = NNODES / 16;              // 6250 (exact)
  const int  waveTiles4 = mtiles * (HID / 64);      // 25000 16x64 wave-tiles
  const long aggTotal   = (long)NNODES * HID;
  const long aggThreads = (long)NEDGES * 32;
  const int  gemmBlocks = blocks((long)waveTiles4 * 32, BT);

  // ---- layer 1 ----
  gemm4_kernel<<<gemmBlocks, BT, 0, stream>>>(Abf, K1PAD, W1p, H, K1PAD, HID, HID/16, waveTiles4);
  fill_zero_f32<<<blocks(aggTotal, BT), BT, 0, stream>>>(AGG, aggTotal);
  agg_kernel<<<blocks(aggThreads, BT), BT, 0, stream>>>(esrc, edst, H, AGG, NEDGES);
  post_kernel<<<blocks(aggTotal, BT), BT, 0, stream>>>(AGG, inn, outn, b1, Abf, 1, aggTotal);

  // ---- layer 2 ----
  gemm4_kernel<<<gemmBlocks, BT, 0, stream>>>(Abf, HID, W2p, H, HID, HID, HID/16, waveTiles4);
  fill_zero_f32<<<blocks(aggTotal, BT), BT, 0, stream>>>(AGG, aggTotal);
  agg_kernel<<<blocks(aggThreads, BT), BT, 0, stream>>>(esrc, edst, H, AGG, NEDGES);
  post_kernel<<<blocks(aggTotal, BT), BT, 0, stream>>>(AGG, inn, outn, b2, Abf, 1, aggTotal);

  // ---- layer 3 ---- (last conv: no out_norm scaling on the activation)
  gemm4_kernel<<<gemmBlocks, BT, 0, stream>>>(Abf, HID, W3p, H, HID, HID, HID/16, waveTiles4);
  fill_zero_f32<<<blocks(aggTotal, BT), BT, 0, stream>>>(AGG, aggTotal);
  agg_kernel<<<blocks(aggThreads, BT), BT, 0, stream>>>(esrc, edst, H, AGG, NEDGES);
  post_kernel<<<blocks(aggTotal, BT), BT, 0, stream>>>(AGG, inn, outn, b3, Abf, 0, aggTotal);

  // ---- classifier ----
  gemm_final_kernel<<<blocks((long)mtiles * 32, BT), BT, 0, stream>>>(Abf, HID, Wfp, bfv, out, HID, mtiles);
}
